// GAT_41188736369378
// MI455X (gfx1250) — compile-verified, matrix operand order checked by reading, not run on hardware
//
#include <hip/hip_runtime.h>
#include <hip/hip_bf16.h>
#include <stdint.h>

typedef __bf16 bf16;
typedef __attribute__((ext_vector_type(16))) __bf16 v16bf;
typedef __attribute__((ext_vector_type(8)))  __bf16 v8bf;
typedef __attribute__((ext_vector_type(8)))  float  v8f;

#define H1v 8
#define C1v 48
#define HIDv 384
#define NEG_SLOPE 0.2f

// ------------------------- helpers -------------------------

__device__ __forceinline__ bf16 f2bf(float f) {
  union { float f; uint32_t u; } v; v.f = f;
  uint32_t r = v.u + 0x7FFFu + ((v.u >> 16) & 1u);   // round-to-nearest-even
  uint16_t h = (uint16_t)(r >> 16);
  return __builtin_bit_cast(bf16, h);
}

__device__ __forceinline__ void edge_sd(const int* __restrict__ ei, int e, int nE,
                                        int& s, int& d) {
  if (e < nE) { s = ei[e]; d = ei[nE + e]; }
  else        { s = e - nE; d = e - nE; }      // self loops appended
}

__device__ __forceinline__ void atomicMaxFloat(float* addr, float val) {
  // classic signed/unsigned ordering trick (init must be -inf)
  if (val >= 0.0f) atomicMax((int*)addr, __float_as_int(val));
  else             atomicMin((unsigned int*)addr, __float_as_uint(val));
}

// ------------------------- utility kernels -------------------------

__global__ void fill_f32(float* __restrict__ p, float v, int n) {
  int t = blockIdx.x * blockDim.x + threadIdx.x;
  if (t < n) p[t] = v;
}

__global__ void cvt_f32_bf16(const float* __restrict__ s, bf16* __restrict__ d, int n) {
  int t = blockIdx.x * blockDim.x + threadIdx.x;
  if (t < n) d[t] = f2bf(s[t]);
}

// ------------------------- weight pre-swizzle -------------------------
// Pack W[kin x 384] (f32, row-major) into WMMA B-fragment-major bf16:
// fragment f = ks*24 + j covers K[ks*32..+32) x N[j*16..+16).
// Lane layout per ISA: lane = half*16 + n15; element i -> k = (i<8?i:i+8)+half*8.
// Storage: Wsw[(f*32 + lane)*16 + i] -> each lane's 16 bf16 contiguous (32B),
// so a wave fetches one fragment as two coalesced global_load_b128 per lane.
__global__ void swizzle_W_f32_bf16(const float* __restrict__ W,
                                   bf16* __restrict__ Wsw, int kin) {
  int t = blockIdx.x * blockDim.x + threadIdx.x;
  int nthread = (kin >> 5) * 24 * 32;
  if (t >= nthread) return;
  int lane = t & 31;
  int f    = t >> 5;            // ks*24 + j
  int ks   = f / 24;
  int j    = f - ks * 24;
  int half = lane >> 4, l15 = lane & 15;
  int ncol = j * 16 + l15;
  bf16* dst = Wsw + (size_t)t * 16;
#pragma unroll
  for (int i = 0; i < 16; ++i) {
    int k = ks * 32 + (i < 8 ? i : i + 8) + half * 8;
    dst[i] = f2bf(W[(size_t)k * HIDv + ncol]);
  }
}

// ------------------------- WMMA GEMM -------------------------
// Y[n x 384] = X[n x kin](bf16) @ W(swizzled bf16) + bias, fp32 out.
// Block: 256 thr = 8 waves; block tile 64 x 384; wave tile 16 x 192 (12 accum).
// X tile double-buffered in LDS (one barrier per k-step); B fragments come
// straight from the pre-swizzled L2-resident weight buffer and are register
// double-buffered so each WMMA overlaps the next fragment's loads.
#define LDS_PITCH 40   // bf16 per LDS row (80B) -> 16B-aligned fragment loads

__device__ __forceinline__ v16bf load_frag(const bf16* __restrict__ p) {
  v8bf lo = *(const v8bf*)(p);
  v8bf hi = *(const v8bf*)(p + 8);
  return __builtin_shufflevector(lo, hi, 0,1,2,3,4,5,6,7,8,9,10,11,12,13,14,15);
}

__global__ __launch_bounds__(256) void gemm_bf16_wmma(
    const bf16* __restrict__ X, const bf16* __restrict__ Wsw,
    const float* __restrict__ bias, float* __restrict__ Y,
    int n, int kin)
{
  __shared__ bf16 lds_x[2][64 * LDS_PITCH];   // 2 x 5 KB ping-pong

  const int tid    = threadIdx.x;
  const int wid    = tid >> 5;
  const int lane   = tid & 31;
  const int m_tile = wid >> 1;      // 0..3
  const int n_half = wid & 1;       // 0..1
  const int half   = lane >> 4;
  const int l15    = lane & 15;
  const int m0     = blockIdx.x * 64;

  v8f acc[12];
#pragma unroll
  for (int j = 0; j < 12; ++j)
    acc[j] = (v8f){0.f,0.f,0.f,0.f,0.f,0.f,0.f,0.f};

  const int ksteps = kin >> 5;
  for (int ks = 0; ks < ksteps; ++ks) {
    bf16* lx = lds_x[ks & 1];
    // --- stage X tile: 64 rows x 32 k (1024 dwords, 4 per thread) ---
#pragma unroll
    for (int i = 0; i < 4; ++i) {
      int idx = tid * 4 + i;          // 0..1023
      int row = idx >> 4;
      int kp  = idx & 15;             // k-pair
      uint32_t v = 0;
      int gr = m0 + row;
      if (gr < n)
        v = *(const uint32_t*)(X + (size_t)gr * kin + ks * 32 + kp * 2);
      *(uint32_t*)(&lx[row * LDS_PITCH + kp * 2]) = v;
    }
    // prefetch next k-step's X rows while WMMAs run
    if (ks + 1 < ksteps) {
      int row = tid >> 2;
      int gr  = m0 + row;
      if (gr < n)
        __builtin_prefetch((const void*)(X + (size_t)gr * kin + (ks + 1) * 32), 0, 1);
    }
    __syncthreads();

    // --- A fragment: lane<16 holds K{0..7,16..23}, lane>=16 K{8..15,24..31} ---
    const bf16* ax = &lx[(m_tile * 16 + l15) * LDS_PITCH + half * 8];
    v8bf alo = *(const v8bf*)(ax);
    v8bf ahi = *(const v8bf*)(ax + 16);
    v16bf a  = __builtin_shufflevector(alo, ahi, 0,1,2,3,4,5,6,7,8,9,10,11,12,13,14,15);

    // --- B fragments: register double-buffered direct loads (L2-resident) ---
    const bf16* bbase = Wsw + ((size_t)((ks * 24 + n_half * 12) * 32) + lane) * 16;
    v16bf bcur = load_frag(bbase);
#pragma unroll
    for (int j = 0; j < 11; ++j) {
      v16bf bnext = load_frag(bbase + (size_t)(j + 1) * (32 * 16));
      acc[j] = __builtin_amdgcn_wmma_f32_16x16x32_bf16(
          false, a, false, bcur, (short)0, acc[j], false, false);
      bcur = bnext;
    }
    acc[11] = __builtin_amdgcn_wmma_f32_16x16x32_bf16(
        false, a, false, bcur, (short)0, acc[11], false, false);
  }

  // --- store: VGPR r -> row (half*8 + r), col = lane&15 within each 16-tile ---
#pragma unroll
  for (int j = 0; j < 12; ++j) {
    int col = n_half * 192 + j * 16 + l15;
    float bv = bias[col];
#pragma unroll
    for (int r = 0; r < 8; ++r) {
      int row = m0 + m_tile * 16 + half * 8 + r;
      if (row < n) Y[(size_t)row * HIDv + col] = acc[j][r] + bv;
    }
  }
}

// ------------------------- edge-phase kernels (layers 1,2) -------------------------

__global__ void edge_score(const float* __restrict__ xl, const float* __restrict__ xr,
                           const int* __restrict__ ei, const float* __restrict__ att,
                           float* __restrict__ score, float* __restrict__ smax,
                           int nE, int nTot)
{
  int t = blockIdx.x * blockDim.x + threadIdx.x;
  if (t >= nTot * H1v) return;
  int e = t >> 3, h = t & 7;
  int s, d; edge_sd(ei, e, nE, s, d);
  const float* pl = xl + (size_t)s * HIDv + h * C1v;
  const float* pr = xr + (size_t)d * HIDv + h * C1v;
  const float* pa = att + h * C1v;
  float acc = 0.f;
#pragma unroll 8
  for (int c = 0; c < C1v; ++c) {
    float m = pl[c] + pr[c];
    m = (m > 0.f) ? m : NEG_SLOPE * m;
    acc += m * pa[c];
  }
  score[(size_t)e * H1v + h] = acc;
  atomicMaxFloat(&smax[(size_t)d * H1v + h], acc);
}

__global__ void edge_exp(const int* __restrict__ ei, float* __restrict__ score,
                         const float* __restrict__ smax, float* __restrict__ denom,
                         int nE, int nTot)
{
  int t = blockIdx.x * blockDim.x + threadIdx.x;
  if (t >= nTot * H1v) return;
  int e = t >> 3, h = t & 7;
  int s, d; edge_sd(ei, e, nE, s, d);
  float ex = __expf(score[(size_t)e * H1v + h] - smax[(size_t)d * H1v + h]);
  score[(size_t)e * H1v + h] = ex;
  atomicAdd(&denom[(size_t)d * H1v + h], ex);
}

__global__ __launch_bounds__(128) void edge_scatter(
    const int* __restrict__ ei, const float* __restrict__ xl,
    const float* __restrict__ ex, const float* __restrict__ denom,
    float* __restrict__ acc, int nE, int nTot)
{
  int e = blockIdx.x;
  if (e >= nTot) return;
  int s, d; edge_sd(ei, e, nE, s, d);
  const float* src = xl + (size_t)s * HIDv;
  float* dst = acc + (size_t)d * HIDv;
#pragma unroll
  for (int i = 0; i < 3; ++i) {
    int c = threadIdx.x + i * 128;
    int h = c / C1v;
    float alpha = ex[(size_t)e * H1v + h] / (denom[(size_t)d * H1v + h] + 1e-16f);
    atomicAdd(&dst[c], src[c] * alpha);
  }
}

__global__ void elu_bias_cvt(const float* __restrict__ acc, const float* __restrict__ bias,
                             float* __restrict__ h, bf16* __restrict__ hbf, int total)
{
  int t = blockIdx.x * blockDim.x + threadIdx.x;
  if (t >= total) return;
  float v = acc[t] + bias[t % HIDv];
  v = (v > 0.f) ? v : (__expf(v) - 1.f);
  h[t] = v;
  hbf[t] = f2bf(v);
}

// ------------------------- layer 3 (384 -> 2, 1 head) -------------------------

__global__ void gemm3(const float* __restrict__ h,
                      const float* __restrict__ Wl, const float* __restrict__ bl,
                      const float* __restrict__ Wr, const float* __restrict__ br,
                      float* __restrict__ xl, float* __restrict__ xr, int n)
{
  int node = blockIdx.x * blockDim.x + threadIdx.x;
  if (node >= n) return;
  const float* hp = h + (size_t)node * HIDv;
  float a0 = bl[0], a1 = bl[1], b0 = br[0], b1 = br[1];
  for (int c = 0; c < HIDv; ++c) {
    float v = hp[c];
    a0 += v * Wl[c * 2];  a1 += v * Wl[c * 2 + 1];
    b0 += v * Wr[c * 2];  b1 += v * Wr[c * 2 + 1];
  }
  xl[node * 2] = a0; xl[node * 2 + 1] = a1;
  xr[node * 2] = b0; xr[node * 2 + 1] = b1;
}

__global__ void edge_score3(const float* __restrict__ xl, const float* __restrict__ xr,
                            const int* __restrict__ ei, const float* __restrict__ att,
                            float* __restrict__ score, float* __restrict__ smax,
                            int nE, int nTot)
{
  int e = blockIdx.x * blockDim.x + threadIdx.x;
  if (e >= nTot) return;
  int s, d; edge_sd(ei, e, nE, s, d);
  float m0 = xl[s * 2]     + xr[d * 2];
  float m1 = xl[s * 2 + 1] + xr[d * 2 + 1];
  m0 = (m0 > 0.f) ? m0 : NEG_SLOPE * m0;
  m1 = (m1 > 0.f) ? m1 : NEG_SLOPE * m1;
  float sc = m0 * att[0] + m1 * att[1];
  score[e] = sc;
  atomicMaxFloat(&smax[d], sc);
}

__global__ void edge_exp3(const int* __restrict__ ei, float* __restrict__ score,
                          const float* __restrict__ smax, float* __restrict__ denom,
                          int nE, int nTot)
{
  int e = blockIdx.x * blockDim.x + threadIdx.x;
  if (e >= nTot) return;
  int s, d; edge_sd(ei, e, nE, s, d);
  float ex = __expf(score[e] - smax[d]);
  score[e] = ex;
  atomicAdd(&denom[d], ex);
}

__global__ void edge_scatter3(const int* __restrict__ ei, const float* __restrict__ xl,
                              const float* __restrict__ ex, const float* __restrict__ denom,
                              float* __restrict__ acc, int nE, int nTot)
{
  int e = blockIdx.x * blockDim.x + threadIdx.x;
  if (e >= nTot) return;
  int s, d; edge_sd(ei, e, nE, s, d);
  float alpha = ex[e] / (denom[d] + 1e-16f);
  atomicAdd(&acc[d * 2],     xl[s * 2]     * alpha);
  atomicAdd(&acc[d * 2 + 1], xl[s * 2 + 1] * alpha);
}

__global__ void final_lsm(const float* __restrict__ acc, const float* __restrict__ bias,
                          float* __restrict__ out, int n)
{
  int t = blockIdx.x * blockDim.x + threadIdx.x;
  if (t >= n) return;
  float v0 = acc[t * 2]     + bias[0];
  float v1 = acc[t * 2 + 1] + bias[1];
  float m  = fmaxf(v0, v1);
  float lse = m + __logf(__expf(v0 - m) + __expf(v1 - m));
  out[t * 2]     = v0 - lse;
  out[t * 2 + 1] = v1 - lse;
}

// ------------------------- host orchestration -------------------------

extern "C" void kernel_launch(void* const* d_in, const int* in_sizes, int n_in,
                              void* d_out, int out_size, void* d_ws, size_t ws_size,
                              hipStream_t stream) {
  (void)n_in; (void)out_size; (void)ws_size;
  const float* x     = (const float*)d_in[0];
  const int*   ei    = (const int*)  d_in[1];
  const float* Wl[2]   = {(const float*)d_in[2],  (const float*)d_in[8]};
  const float* blv[2]  = {(const float*)d_in[3],  (const float*)d_in[9]};
  const float* Wr[2]   = {(const float*)d_in[4],  (const float*)d_in[10]};
  const float* brv[2]  = {(const float*)d_in[5],  (const float*)d_in[11]};
  const float* attv[2] = {(const float*)d_in[6],  (const float*)d_in[12]};
  const float* biasv[2]= {(const float*)d_in[7],  (const float*)d_in[13]};
  const float* Wl3 = (const float*)d_in[14]; const float* bl3 = (const float*)d_in[15];
  const float* Wr3 = (const float*)d_in[16]; const float* br3 = (const float*)d_in[17];
  const float* att3= (const float*)d_in[18]; const float* bias3=(const float*)d_in[19];

  const int N    = in_sizes[0] / 128;
  const int E    = in_sizes[1] / 2;
  const int Etot = E + N;
  const int kins[2] = {128, HIDv};

  // --- carve workspace ---
  char* wsb = (char*)d_ws;
  size_t off = 0;
  auto carve = [&](size_t bytes) -> void* {
    void* p = wsb + off;
    off += (bytes + 255) & ~(size_t)255;
    return p;
  };
  float* xlb   = (float*)carve((size_t)N * HIDv * 4);
  float* xrb   = (float*)carve((size_t)N * HIDv * 4);
  float* accb  = (float*)carve((size_t)N * HIDv * 4);
  float* hbuf  = (float*)carve((size_t)N * HIDv * 4);
  bf16*  xbf   = (bf16*) carve((size_t)N * HIDv * 2);
  bf16*  wbfL  = (bf16*) carve((size_t)HIDv * HIDv * 2);
  bf16*  wbfR  = (bf16*) carve((size_t)HIDv * HIDv * 2);
  float* score = (float*)carve((size_t)Etot * H1v * 4);
  float* smax  = (float*)carve((size_t)N * H1v * 4);
  float* denom = (float*)carve((size_t)N * H1v * 4);

  const float ninf = -__builtin_inff();
  auto cdiv = [](int a, int b) { return (a + b - 1) / b; };

  // ---- layers 1 & 2 ----
  for (int l = 0; l < 2; ++l) {
    const int kin = kins[l];
    if (l == 0)
      cvt_f32_bf16<<<cdiv(N * kin, 256), 256, 0, stream>>>(x, xbf, N * kin);
    // fused convert + swizzle of weights into B-fragment-major layout
    const int swt = kin * 24;   // (kin/32) * 24 frags * 32 lanes
    swizzle_W_f32_bf16<<<cdiv(swt, 256), 256, 0, stream>>>(Wl[l], wbfL, kin);
    swizzle_W_f32_bf16<<<cdiv(swt, 256), 256, 0, stream>>>(Wr[l], wbfR, kin);

    const int gb = cdiv(N, 64);
    gemm_bf16_wmma<<<gb, 256, 0, stream>>>(xbf, wbfL, blv[l], xlb, N, kin);
    gemm_bf16_wmma<<<gb, 256, 0, stream>>>(xbf, wbfR, brv[l], xrb, N, kin);

    fill_f32<<<cdiv(N * H1v, 256), 256, 0, stream>>>(smax, ninf, N * H1v);
    fill_f32<<<cdiv(N * H1v, 256), 256, 0, stream>>>(denom, 0.f, N * H1v);
    fill_f32<<<cdiv(N * HIDv, 256), 256, 0, stream>>>(accb, 0.f, N * HIDv);

    edge_score<<<cdiv(Etot * H1v, 256), 256, 0, stream>>>(xlb, xrb, ei, attv[l],
                                                          score, smax, E, Etot);
    edge_exp<<<cdiv(Etot * H1v, 256), 256, 0, stream>>>(ei, score, smax, denom, E, Etot);
    edge_scatter<<<Etot, 128, 0, stream>>>(ei, xlb, score, denom, accb, E, Etot);

    elu_bias_cvt<<<cdiv(N * HIDv, 256), 256, 0, stream>>>(accb, biasv[l], hbuf, xbf,
                                                          N * HIDv);
  }

  // ---- layer 3 ----
  gemm3<<<cdiv(N, 256), 256, 0, stream>>>(hbuf, Wl3, bl3, Wr3, br3, xlb, xrb, N);
  fill_f32<<<cdiv(N, 256), 256, 0, stream>>>(smax, ninf, N);
  fill_f32<<<cdiv(N, 256), 256, 0, stream>>>(denom, 0.f, N);
  fill_f32<<<cdiv(N * 2, 256), 256, 0, stream>>>(accb, 0.f, N * 2);

  edge_score3<<<cdiv(Etot, 256), 256, 0, stream>>>(xlb, xrb, ei, att3, score, smax, E, Etot);
  edge_exp3<<<cdiv(Etot, 256), 256, 0, stream>>>(ei, score, smax, denom, E, Etot);
  edge_scatter3<<<cdiv(Etot, 256), 256, 0, stream>>>(ei, xlb, score, denom, accb, E, Etot);

  final_lsm<<<cdiv(N, 256), 256, 0, stream>>>(accb, bias3, (float*)d_out, N);
}
